// RUMLayer_85057532330620
// MI455X (gfx1250) — matrix operand order repaired and checked by reading, not compiled, not run on hardware
//
#include <hip/hip_runtime.h>
#include <math.h>

#define LSEQ 8
#define HDIM 64
#define LPAD 72   // f16 row stride for LDS tiles (bank-spread, 16B aligned)

typedef __attribute__((ext_vector_type(16))) _Float16 v16h;
typedef __attribute__((ext_vector_type(8)))  float    v8f;

static __device__ __forceinline__ v8f wmma16(v16h a, v16h b, v8f c) {
  // D = A(16x32 f16) * B(32x16 f16) + C(16x16 f32)
  return __builtin_amdgcn_wmma_f32_16x16x32_f16(false, a, false, b, (short)0, c, false, false);
}

static __device__ __forceinline__ v8f vzero8() {
  v8f z;
#pragma unroll
  for (int i = 0; i < 8; ++i) z[i] = 0.0f;
  return z;
}

// fast sigmoid: v_exp_f32 + v_rcp_f32 (gate preactivations are O(1), saturating)
static __device__ __forceinline__ float sigmf(float x) {
  return __builtin_amdgcn_rcpf(1.0f + __expf(-x));
}

// fast tanh: tanh(x) = (1 - e^{-2x}) / (1 + e^{-2x}), clamped to avoid overflow
static __device__ __forceinline__ float tanh_fast(float x) {
  const float xc = fminf(15.0f, fmaxf(-15.0f, x));
  const float e  = __expf(-2.0f * xc);
  return (1.0f - e) * __builtin_amdgcn_rcpf(1.0f + e);
}

// A fragment (16x32, f16) from LDS row-major [16][LPAD]; kc = K-chunk base (0|32).
// 16-bit A layout: lane holds row m=lane&15; K runs {kb..kb+7, kb+16..kb+23}, kb = kc + 8*(lane>>4).
static __device__ __forceinline__ v16h ldsA(const _Float16 (*buf)[LPAD], int lane, int kc) {
  const int m  = lane & 15;
  const int kb = kc + ((lane >> 4) & 1) * 8;
  const _Float16* p = &buf[m][kb];
  v16h a;
#pragma unroll
  for (int i = 0; i < 8; ++i) { a[i] = p[i]; a[8 + i] = p[16 + i]; }
  return a;
}

// A fragment from a global f32 row (per-lane row base = hglob + node(m)*64), converted to f16.
static __device__ __forceinline__ v16h ldAg(const float* base, int lane, int kc) {
  const float* p = base + kc + ((lane >> 4) & 1) * 8;
  v16h a;
#pragma unroll
  for (int i = 0; i < 8; ++i) { a[i] = (_Float16)p[i]; a[8 + i] = (_Float16)p[16 + i]; }
  return a;
}

// B fragment (32x16, f16) of W^T from row-major W[rows][ld] (f32 -> f16).
// B layout: lane holds col n = nbase + (lane&15); K = 8*(lane>>4) + (i&7) + 16*(i>>3) (+kbase).
static __device__ __forceinline__ v16h ldB(const float* W, int ld, int nbase, int kbase, int lane) {
  const float* p = W + (nbase + (lane & 15)) * ld + kbase + ((lane >> 4) & 1) * 8;
  v16h b;
#pragma unroll
  for (int i = 0; i < 16; ++i) b[i] = (_Float16)p[(i & 7) + 16 * (i >> 3)];
  return b;
}

// One direction of the positional-encoding GRU (input dim 2 on VALU,
// recurrent 16x64x192 per step on WMMA). Each wave owns hidden cols [jb, jb+16).
// hbuf is double-buffered: one barrier per recurrence step.
static __device__ void gru_small_stage(
    const float* __restrict__ Whh, const float* __restrict__ Wih2,
    const float* __restrict__ bihv, const float* __restrict__ bhhv,
    _Float16 (*hbuf)[16][LPAD], _Float16 (*yout)[16][LPAD],
    const float (*sb)[LSEQ], const float (*cb)[LSEQ],
    int lane, int jb, bool rev, v8f* hout)
{
  const int nl  = lane & 15;
  const int col = jb + nl;

  // h0 = 0 in buffer 0
  for (int idx = threadIdx.x; idx < 16 * LPAD; idx += blockDim.x)
    (&hbuf[0][0][0])[idx] = (_Float16)0.0f;
  __syncthreads();

  // Whh^T tiles resident in VGPRs (r,z,n gates x 2 K-chunks)
  v16h bR0 = ldB(Whh, HDIM,       jb, 0, lane), bR1 = ldB(Whh, HDIM,       jb, 32, lane);
  v16h bZ0 = ldB(Whh, HDIM,  64 + jb, 0, lane), bZ1 = ldB(Whh, HDIM,  64 + jb, 32, lane);
  v16h bN0 = ldB(Whh, HDIM, 128 + jb, 0, lane), bN1 = ldB(Whh, HDIM, 128 + jb, 32, lane);

  // per-lane input weights / fused biases (input dim = 2: sin, cos)
  const float wr0 = Wih2[(      col) * 2], wr1 = Wih2[(      col) * 2 + 1];
  const float wz0 = Wih2[( 64 + col) * 2], wz1 = Wih2[( 64 + col) * 2 + 1];
  const float wn0 = Wih2[(128 + col) * 2], wn1 = Wih2[(128 + col) * 2 + 1];
  const float brt = bihv[col]       + bhhv[col];        // r: x-bias + h-bias
  const float bzt = bihv[64 + col]  + bhhv[64 + col];   // z
  const float bnx = bihv[128 + col];                    // n: x-bias (h-bias gated by r)
  const float bnh = bhhv[128 + col];

  v8f h = vzero8();
  for (int s = 0; s < LSEQ; ++s) {
    const int t   = rev ? (LSEQ - 1 - s) : s;
    const int cur = s & 1, nxt = cur ^ 1;
    v16h a0 = ldsA(hbuf[cur], lane, 0);
    v16h a1 = ldsA(hbuf[cur], lane, 32);
    v8f aR = vzero8(), aZ = vzero8(), aN = vzero8();
    aR = wmma16(a0, bR0, aR); aR = wmma16(a1, bR1, aR);
    aZ = wmma16(a0, bZ0, aZ); aZ = wmma16(a1, bZ1, aZ);
    aN = wmma16(a0, bN0, aN); aN = wmma16(a1, bN1, aN);
#pragma unroll
    for (int v = 0; v < 8; ++v) {
      const int m = v + 8 * ((lane >> 4) & 1);   // C-layout row
      const float si = sb[m][t], co = cb[m][t];
      const float r  = sigmf(fmaf(wr0, si, fmaf(wr1, co, brt)) + aR[v]);
      const float z  = sigmf(fmaf(wz0, si, fmaf(wz1, co, bzt)) + aZ[v]);
      const float nn = tanh_fast(fmaf(wn0, si, fmaf(wn1, co, bnx)) + r * (aN[v] + bnh));
      const float hv = fmaf(z, h[v] - nn, nn);   // (1-z)*n + z*h
      h[v] = hv;
      const _Float16 hh = (_Float16)hv;
      hbuf[nxt][m][col] = hh;                    // write next-step state (no WAR on cur)
      yout[t][m][col]   = hh;
    }
    __syncthreads();                             // single barrier per step
  }
  *hout = h;
}

__global__ __launch_bounds__(128) void rum_fused_kernel(
    const float* __restrict__ hglob, const int* __restrict__ walks,
    const int* __restrict__ indeg,
    const float* __restrict__ Wih_f, const float* __restrict__ Whh_f,
    const float* __restrict__ bih_f, const float* __restrict__ bhh_f,
    const float* __restrict__ Wih_b, const float* __restrict__ Whh_b,
    const float* __restrict__ bih_b, const float* __restrict__ bhh_b,
    const float* __restrict__ Wih,   const float* __restrict__ Whh,
    const float* __restrict__ bih,   const float* __restrict__ bhh,
    float* __restrict__ out, const int* __restrict__ ws)
{
  __shared__ _Float16 hbuf[2][16][LPAD];       // double-buffered hidden state
  __shared__ _Float16 yf[LSEQ][16][LPAD];
  __shared__ _Float16 yb[LSEQ][16][LPAD];
  __shared__ int   worig[16][LSEQ];
  __shared__ int   wrev [16][LSEQ];
  __shared__ float sbuf[16][LSEQ], cbuf[16][LSEQ], dbuf[16][LSEQ];

  const int tid  = threadIdx.x;
  const int lane = tid & 31;
  const int jb   = (tid >> 5) * 16;   // wave's hidden-column block
  const int nl   = lane & 15;
  const int b0   = blockIdx.x * 16;   // first walk of this tile

  __builtin_prefetch(Wih, 0, 3);
  __builtin_prefetch(Whh, 0, 3);

  // ---- walk preprocessing: reversal, first-occurrence sin/cos, normalized degree ----
  {
    const int m = tid >> 3, i = tid & 7;
    worig[m][i] = walks[(b0 + m) * LSEQ + i];
  }
  __syncthreads();
  {
    const int m = tid >> 3, i = tid & 7;
    const int wi = worig[m][i];
    int j = i;
#pragma unroll
    for (int jj = LSEQ - 1; jj >= 0; --jj)
      if (worig[m][jj] == wi) j = jj;          // smallest matching index == argmax of equality
    const int ip = LSEQ - 1 - i;               // reversed position
    wrev[m][ip] = wi;
    const float u = (float)j * (6.2831853071795864f / (float)LSEQ);
    sbuf[m][ip] = __sinf(u);
    cbuf[m][ip] = __cosf(u);
    dbuf[m][ip] = (float)indeg[wi] * __builtin_amdgcn_rcpf((float)ws[0]);
  }
  __syncthreads();

  // ---- fwd / bwd positional GRUs ----
  v8f hf, hbk;
  gru_small_stage(Whh_f, Wih_f, bih_f, bhh_f, hbuf, yf, sbuf, cbuf, lane, jb, false, &hf);
  gru_small_stage(Whh_b, Wih_b, bih_b, bhh_b, hbuf, yb, sbuf, cbuf, lane, jb, true,  &hbk);

  // ---- final GRU: x = [h[node] | yf,yb | deg], h0 = 0.5*(hf+hb) ----
  v8f h = vzero8();
#pragma unroll
  for (int v = 0; v < 8; ++v) {
    h[v] = 0.5f * (hf[v] + hbk[v]);
    const int m = v + 8 * ((lane >> 4) & 1);
    hbuf[0][m][jb + nl] = (_Float16)h[v];      // waves write disjoint column blocks
  }
  __syncthreads();

  const int LD = 2 * HDIM + HDIM + 1;          // 193
  // recurrent Whh^T tiles
  v16h bhR0 = ldB(Whh, HDIM,       jb, 0, lane), bhR1 = ldB(Whh, HDIM,       jb, 32, lane);
  v16h bhZ0 = ldB(Whh, HDIM,  64 + jb, 0, lane), bhZ1 = ldB(Whh, HDIM,  64 + jb, 32, lane);
  v16h bhN0 = ldB(Whh, HDIM, 128 + jb, 0, lane), bhN1 = ldB(Whh, HDIM, 128 + jb, 32, lane);
  // input Wih^T tiles: node-feature part (cols 0..63)
  v16h bxR0 = ldB(Wih, LD,       jb, 0, lane), bxR1 = ldB(Wih, LD,       jb, 32, lane);
  v16h bxZ0 = ldB(Wih, LD,  64 + jb, 0, lane), bxZ1 = ldB(Wih, LD,  64 + jb, 32, lane);
  v16h bxN0 = ldB(Wih, LD, 128 + jb, 0, lane), bxN1 = ldB(Wih, LD, 128 + jb, 32, lane);
  // input Wih^T tiles: y_walk part (cols 64..191)
  v16h byR[4], byZ[4], byN[4];
#pragma unroll
  for (int c = 0; c < 4; ++c) {
    byR[c] = ldB(Wih, LD,       jb, 64 + 32 * c, lane);
    byZ[c] = ldB(Wih, LD,  64 + jb, 64 + 32 * c, lane);
    byN[c] = ldB(Wih, LD, 128 + jb, 64 + 32 * c, lane);
  }
  const int colg = jb + nl;
  const float fbrt = bih[colg]       + bhh[colg];         // fused r bias
  const float fbzt = bih[64 + colg]  + bhh[64 + colg];    // fused z bias
  const float fbnx = bih[128 + colg];
  const float fbnh = bhh[128 + colg];
  const float wdr = Wih[(      colg) * LD + LD - 1];
  const float wdz = Wih[( 64 + colg) * LD + LD - 1];
  const float wdn = Wih[(128 + colg) * LD + LD - 1];

  for (int t = 0; t < LSEQ; ++t) {
    const int cur = t & 1, nxt = cur ^ 1;
    // A fragments: recurrent h, gathered node features, y_walk
    v16h ah0 = ldsA(hbuf[cur], lane, 0);
    v16h ah1 = ldsA(hbuf[cur], lane, 32);
    const int node = wrev[nl][t];              // A row m == lane&15
    const float* hp = hglob + node * HDIM;
    v16h ax0 = ldAg(hp, lane, 0);
    v16h ax1 = ldAg(hp, lane, 32);
    v16h ay0 = ldsA(yf[t], lane, 0);
    v16h ay1 = ldsA(yf[t], lane, 32);
    v16h ay2 = ldsA(yb[t], lane, 0);
    v16h ay3 = ldsA(yb[t], lane, 32);

    v8f aR = vzero8(), aZ = vzero8(), aXN = vzero8(), aHN = vzero8();
    aR  = wmma16(ah0, bhR0, aR);  aR  = wmma16(ah1, bhR1, aR);
    aZ  = wmma16(ah0, bhZ0, aZ);  aZ  = wmma16(ah1, bhZ1, aZ);
    aHN = wmma16(ah0, bhN0, aHN); aHN = wmma16(ah1, bhN1, aHN);
    aR  = wmma16(ax0, bxR0, aR);  aR  = wmma16(ax1, bxR1, aR);
    aZ  = wmma16(ax0, bxZ0, aZ);  aZ  = wmma16(ax1, bxZ1, aZ);
    aXN = wmma16(ax0, bxN0, aXN); aXN = wmma16(ax1, bxN1, aXN);
    aR  = wmma16(ay0, byR[0], aR);  aR  = wmma16(ay1, byR[1], aR);
    aR  = wmma16(ay2, byR[2], aR);  aR  = wmma16(ay3, byR[3], aR);
    aZ  = wmma16(ay0, byZ[0], aZ);  aZ  = wmma16(ay1, byZ[1], aZ);
    aZ  = wmma16(ay2, byZ[2], aZ);  aZ  = wmma16(ay3, byZ[3], aZ);
    aXN = wmma16(ay0, byN[0], aXN); aXN = wmma16(ay1, byN[1], aXN);
    aXN = wmma16(ay2, byN[2], aXN); aXN = wmma16(ay3, byN[3], aXN);
#pragma unroll
    for (int v = 0; v < 8; ++v) {
      const int m  = v + 8 * ((lane >> 4) & 1);
      const float dv = dbuf[m][t];
      const float r  = sigmf(aR[v] + fmaf(dv, wdr, fbrt));
      const float z  = sigmf(aZ[v] + fmaf(dv, wdz, fbzt));
      const float nn = tanh_fast(aXN[v] + fmaf(dv, wdn, fbnx) + r * (aHN[v] + fbnh));
      const float hv = fmaf(z, h[v] - nn, nn);
      h[v] = hv;
      hbuf[nxt][m][jb + nl] = (_Float16)hv;    // write next-step buffer (no WAR)
    }
    __syncthreads();                           // single barrier per step
  }

  // final hidden state -> out (S,N,H)
#pragma unroll
  for (int v = 0; v < 8; ++v) {
    const int m = v + 8 * ((lane >> 4) & 1);
    out[(b0 + m) * HDIM + jb + nl] = h[v];
  }
}

__global__ void rum_ws_init(int* ws) {
  if (threadIdx.x == 0 && blockIdx.x == 0) ws[0] = 1;   // degrees are >= 1
}

__global__ void rum_degmax(const int* __restrict__ walks, const int* __restrict__ indeg,
                           int n, int* __restrict__ ws) {
  __shared__ int red[256];
  const int i = blockIdx.x * blockDim.x + threadIdx.x;
  int d = 1;
  if (i < n) d = indeg[walks[i]];
  red[threadIdx.x] = d;
  __syncthreads();
  for (int s = 128; s > 0; s >>= 1) {
    if (threadIdx.x < s) red[threadIdx.x] = max(red[threadIdx.x], red[threadIdx.x + s]);
    __syncthreads();
  }
  if (threadIdx.x == 0) atomicMax(ws, red[0]);
}

extern "C" void kernel_launch(void* const* d_in, const int* in_sizes, int n_in,
                              void* d_out, int out_size, void* d_ws, size_t ws_size,
                              hipStream_t stream) {
  (void)n_in; (void)out_size; (void)ws_size;
  const float* hglob = (const float*)d_in[0];
  const int*   walks = (const int*)d_in[1];
  const int*   indeg = (const int*)d_in[2];
  const float* Wih_f = (const float*)d_in[3];
  const float* Whh_f = (const float*)d_in[4];
  const float* bih_f = (const float*)d_in[5];
  const float* bhh_f = (const float*)d_in[6];
  const float* Wih_b = (const float*)d_in[7];
  const float* Whh_b = (const float*)d_in[8];
  const float* bih_b = (const float*)d_in[9];
  const float* bhh_b = (const float*)d_in[10];
  const float* Wih   = (const float*)d_in[11];
  const float* Whh   = (const float*)d_in[12];
  const float* bih   = (const float*)d_in[13];
  const float* bhh   = (const float*)d_in[14];
  float* out = (float*)d_out;
  int*   ws  = (int*)d_ws;

  const int nwalk_elems = in_sizes[1];        // S*N*L
  const int nwalks      = nwalk_elems / LSEQ; // S*N (40000, divisible by 16)

  rum_ws_init<<<1, 32, 0, stream>>>(ws);
  rum_degmax<<<(nwalk_elems + 255) / 256, 256, 0, stream>>>(walks, indeg, nwalk_elems, ws);
  rum_fused_kernel<<<nwalks / 16, 128, 0, stream>>>(
      hglob, walks, indeg,
      Wih_f, Whh_f, bih_f, bhh_f,
      Wih_b, Whh_b, bih_b, bhh_b,
      Wih, Whh, bih, bhh, out, ws);
}
